// FMoENaiveGate_1958505087362
// MI455X (gfx1250) — matrix operand order repaired and checked by reading, not compiled, not run on hardware
//
#include <hip/hip_runtime.h>

// MoE naive gate for MI455X (gfx1250, wave32, WMMA).
// gate = inp[T,D] @ W[E,D]^T + b ; top-2 over E ; softmax(2) ; emit (idx, score).
// T=16384, D=4096, E=128. HBM-bound (268MB inp stream @ 23.3TB/s ~= 11.5us);
// bf16 WMMA (16x16x32, f32 accum) keeps matrix math far above the 1.5 PFLOP/s
// needed to stay on the bandwidth roofline; W (2MB) lives in L2/WGP$.

typedef __attribute__((ext_vector_type(16))) __bf16 v16bf;
typedef __attribute__((ext_vector_type(8)))  float  v8f;

#define D_MODEL 4096
#define N_EXP   128
#define TOKENS_PER_BLOCK 64   // 4 token tiles of 16; 8 waves (expert-split x2)

__device__ __forceinline__ void cvt8(v16bf& d, int base, float4 x, float4 y) {
  d[base + 0] = (__bf16)x.x; d[base + 1] = (__bf16)x.y;
  d[base + 2] = (__bf16)x.z; d[base + 3] = (__bf16)x.w;
  d[base + 4] = (__bf16)y.x; d[base + 5] = (__bf16)y.y;
  d[base + 6] = (__bf16)y.z; d[base + 7] = (__bf16)y.w;
}

__global__ __launch_bounds__(256, 1)
void moe_gate_kernel(const float* __restrict__ inp,
                     const float* __restrict__ W,
                     const float* __restrict__ bias,
                     int*   __restrict__ out_idx,
                     float* __restrict__ out_score)
{
  // [token_tile][m][expert], padded 128->129 to avoid LDS bank conflicts
  __shared__ float gates[4][16][129];

  const int tid  = threadIdx.x;
  const int wave = tid >> 5;
  const int lane = tid & 31;
  const int l    = lane & 15;   // M (A/C) or N (B/C) within a 16-wide tile
  const int half = lane >> 4;   // lane group 0/1
  const int tile = wave & 3;    // token tile within block (16 tokens)
  const int eg   = wave >> 2;   // expert group (64 experts each)

  const int tokBase = blockIdx.x * TOKENS_PER_BLOCK + tile * 16;
  const float* __restrict__ Arow = inp + (size_t)(tokBase + l) * D_MODEL;

  v8f acc[4] = {};  // 4 expert tiles of 16x16 f32

  for (int k = 0; k < D_MODEL; k += 32) {
    // ---- A fragment: 16x32 bf16, contiguous k order (same as B):
    // lanes 0-15: K = k..k+7 (V0-3), k+8..k+15 (V4-7)
    // lanes 16-31: K = k+16..k+23,    k+24..k+31
    const int kA0 = k + half * 16;
    const int kA1 = k + half * 16 + 8;
    float4 a0 = *(const float4*)(Arow + kA0);
    float4 a1 = *(const float4*)(Arow + kA0 + 4);
    float4 a2 = *(const float4*)(Arow + kA1);
    float4 a3 = *(const float4*)(Arow + kA1 + 4);
    v16bf afrag;
    cvt8(afrag, 0, a0, a1);
    cvt8(afrag, 8, a2, a3);

    // ---- B fragments: 32x16 bf16, lanes 0-15 hold K=k..k+15 (N=lane),
    // lanes 16-31 hold K=k+16..k+31
    const int kB = k + half * 16;
#pragma unroll
    for (int et = 0; et < 4; ++et) {
      const int expert = (eg * 4 + et) * 16 + l;
      const float* __restrict__ Wp = W + (size_t)expert * D_MODEL + kB;
      float4 b0 = *(const float4*)(Wp + 0);
      float4 b1 = *(const float4*)(Wp + 4);
      float4 b2 = *(const float4*)(Wp + 8);
      float4 b3 = *(const float4*)(Wp + 12);
      v16bf bfrag;
      cvt8(bfrag, 0, b0, b1);
      cvt8(bfrag, 8, b2, b3);
      acc[et] = __builtin_amdgcn_wmma_f32_16x16x32_bf16(
          /*neg_a=*/false, afrag, /*neg_b=*/false, bfrag,
          /*c_mod=*/(short)0, acc[et], /*reuse_a=*/false, /*reuse_b=*/false);
    }
  }

  // ---- scatter C (+bias) to LDS. C/D layout: lane<16 -> M=v, lane>=16 -> M=v+8
#pragma unroll
  for (int et = 0; et < 4; ++et) {
    const int expert = (eg * 4 + et) * 16 + l;
    const float bb = bias[expert];
#pragma unroll
    for (int v = 0; v < 8; ++v) {
      gates[tile][v + half * 8][expert] = acc[et][v] + bb;
    }
  }
  __syncthreads();

  // ---- per-token top-2 + softmax (64 tokens -> first 64 lanes)
  if (tid < TOKENS_PER_BLOCK) {
    const float* g = &gates[tid >> 4][tid & 15][0];
    float v0 = -3.402823466e38f, v1 = -3.402823466e38f;
    int   i0 = 0, i1 = 0;
    for (int e = 0; e < N_EXP; ++e) {
      float v = g[e];                 // ascending scan + strict '>' ==
      if (v > v0) {                   // lowest-index-first on ties (jax top_k)
        v1 = v0; i1 = i0; v0 = v; i0 = e;
      } else if (v > v1) {
        v1 = v; i1 = e;
      }
    }
    const float e1 = __expf(v1 - v0);     // v1 <= v0: stable 2-way softmax
    const float s0 = 1.0f / (1.0f + e1);
    const float s1 = e1 * s0;
    const int tok = blockIdx.x * TOKENS_PER_BLOCK + tid;
    out_idx[tok * 2 + 0]   = i0;
    out_idx[tok * 2 + 1]   = i1;
    out_score[tok * 2 + 0] = s0;
    out_score[tok * 2 + 1] = s1;
  }
}

extern "C" void kernel_launch(void* const* d_in, const int* in_sizes, int n_in,
                              void* d_out, int out_size, void* d_ws, size_t ws_size,
                              hipStream_t stream) {
  const float* inp  = (const float*)d_in[0];   // [T, D]
  const float* W    = (const float*)d_in[1];   // [E, D]
  const float* bias = (const float*)d_in[2];   // [E]
  const int T = in_sizes[0] / D_MODEL;

  int*   out_idx   = (int*)d_out;                      // [T*2] int32 bits
  float* out_score = (float*)d_out + (size_t)T * 2;    // [T,1,2] f32

  dim3 grid(T / TOKENS_PER_BLOCK);  // 256 blocks x 8 waves
  moe_gate_kernel<<<grid, 256, 0, stream>>>(inp, W, bias, out_idx, out_score);
}
